// Attention_12859132084363
// MI455X (gfx1250) — matrix operand order repaired and checked by reading, not compiled, not run on hardware
//
#include <hip/hip_runtime.h>
#include <hip/hip_bf16.h>

typedef __attribute__((ext_vector_type(16))) _Float16 v16h;
typedef __attribute__((ext_vector_type(8)))  _Float16 v8h;
typedef __attribute__((ext_vector_type(4)))  _Float16 v4h;
typedef __attribute__((ext_vector_type(2)))  _Float16 v2h;
typedef __attribute__((ext_vector_type(8)))  float    v8f;
typedef __attribute__((ext_vector_type(4)))  int      v4i;

constexpr int BATCH = 2, SEQ = 2048, DIM = 1024, HEADS = 16, HDIM = 64;

#if defined(__AMDGCN__) && __has_builtin(__builtin_amdgcn_global_load_async_to_lds_b128)
#define HAVE_ASYNC_LDS 1
#endif

__device__ __forceinline__ v16h hcat(v8h lo, v8h hi) {
  union { v16h v; v8h h[2]; } u;
  u.h[0] = lo; u.h[1] = hi;
  return u.v;
}

__device__ __forceinline__ v8f wmma16(v16h a, v16h b, v8f c) {
  // D = A(16x32 f16) x B(32x16 f16) + C(16x16 f32)
  return __builtin_amdgcn_wmma_f32_16x16x32_f16(false, a, false, b,
                                                (short)0, c, false, false);
}

// 16-byte global -> LDS copy; async (ASYNCcnt) on gfx1250 when available.
__device__ __forceinline__ void async_copy16(const _Float16* g, _Float16* l) {
#if defined(HAVE_ASYNC_LDS)
  __builtin_amdgcn_global_load_async_to_lds_b128(
      (__attribute__((address_space(1))) v4i*)g,
      (__attribute__((address_space(3))) v4i*)l, 0, 0);
#else
  *(v8h*)l = *(const v8h*)g;
#endif
}

__device__ __forceinline__ void wait_async() {
#if defined(HAVE_ASYNC_LDS)
#if __has_builtin(__builtin_amdgcn_s_wait_asynccnt)
  __builtin_amdgcn_s_wait_asynccnt(0);
#else
  asm volatile("s_wait_asynccnt 0" ::: "memory");
#endif
#endif
}

// ---------------------------------------------------------------------------
// Kernel 1: Y = X @ W + bias, fused RoPE epilogue, f16 output in attention-
// friendly layouts.  mode 0: N=2048 -> q/k [B,H,S,64]; mode 1: N=1024 -> vT
// [B,H,64,S].  Block = 128 threads (4 waves), tile 128(M) x 64(N), K-step 32;
// each wave owns 32(M) x 64(N) => 8 WMMAs per k-step.
// ---------------------------------------------------------------------------
__global__ __launch_bounds__(128) void proj_rope_kernel(
    const float* __restrict__ X, const float* __restrict__ W,
    const float* __restrict__ bias,
    _Float16* __restrict__ qb, _Float16* __restrict__ kb,
    _Float16* __restrict__ vT, int N, int mode)
{
  __shared__ _Float16 As[128][40];  // [m][k], 80B row stride -> conflict-free
  __shared__ _Float16 Bs[64][40];   // [n][k] = W^T tile

  const int tid  = threadIdx.x;
  const int wv   = tid >> 5;
  const int lane = tid & 31;
  const int half = lane >> 4;
  const int l16  = lane & 15;
  const int m0 = blockIdx.y * 128;
  const int n0 = blockIdx.x * 64;

  v8f acc[2][4] = {};

  for (int k0 = 0; k0 < DIM; k0 += 32) {
    // --- stage A tile (128 rows x 32 K): one row per thread, fp32 -> f16 ---
    {
      const float* src = X + (size_t)(m0 + tid) * DIM + k0;
      #pragma unroll
      for (int c = 0; c < 4; ++c) {
        v8h t;
        #pragma unroll
        for (int i = 0; i < 2; ++i) {
          float4 f = *(const float4*)(src + c*8 + i*4);
          t[i*4+0] = (_Float16)f.x; t[i*4+1] = (_Float16)f.y;
          t[i*4+2] = (_Float16)f.z; t[i*4+3] = (_Float16)f.w;
        }
        *(v8h*)&As[tid][c*8] = t;
      }
    }
    // --- stage B^T tile via register 4x4 transpose: Bs[n][k]=W[k0+k][n0+n] ---
    {
      const int kq = (tid & 7) * 4;
      const int nq = (tid >> 3) * 4;
      float4 r[4];
      #pragma unroll
      for (int i = 0; i < 4; ++i)
        r[i] = *(const float4*)(W + (size_t)(k0 + kq + i) * N + n0 + nq);
      #pragma unroll
      for (int j = 0; j < 4; ++j) {
        v4h t;
        t[0] = (_Float16)(&r[0].x)[j]; t[1] = (_Float16)(&r[1].x)[j];
        t[2] = (_Float16)(&r[2].x)[j]; t[3] = (_Float16)(&r[3].x)[j];
        *(v4h*)&Bs[nq + j][kq] = t;     // 8B ds_store
      }
    }
    __syncthreads();

    // A fragments: lane row = wv*32+mt*16+l16, K = {half*8+0..7, half*8+16..23}
    v16h a[2];
    #pragma unroll
    for (int mt = 0; mt < 2; ++mt)
      a[mt] = hcat(*(const v8h*)&As[wv*32 + mt*16 + l16][half*8],
                   *(const v8h*)&As[wv*32 + mt*16 + l16][half*8 + 16]);
    #pragma unroll
    for (int nt = 0; nt < 4; ++nt) {
      // B fragment: lane col = nt*16+l16, K = {half*8+0..7, half*8+16..23}
      v16h b = hcat(*(const v8h*)&Bs[nt*16 + l16][half*8],
                    *(const v8h*)&Bs[nt*16 + l16][half*8 + 16]);
      acc[0][nt] = wmma16(a[0], b, acc[0][nt]);
      acc[1][nt] = wmma16(a[1], b, acc[1][nt]);
    }
    __syncthreads();
  }

  // --- epilogue: bias (+ RoPE for q/k), scatter to f16 buffers ---
  // C layout: lane col n = n0+nt*16+l16, rows m = m0+wv*32+mt*16+half*8+r
  #pragma unroll
  for (int mt = 0; mt < 2; ++mt) {
    #pragma unroll
    for (int nt = 0; nt < 4; ++nt) {
      const int n  = n0 + nt*16 + l16;
      const float bn = bias[n];
      if (mode == 1) {                    // V: write transposed [B,H,64,S]
        const int h = n >> 6, d = n & 63;
        #pragma unroll
        for (int r = 0; r < 8; ++r) {
          const int m = m0 + wv*32 + mt*16 + half*8 + r;
          const int bi = m >> 11, sp = m & (SEQ - 1);
          vT[(((size_t)bi*HEADS + h)*HDIM + d)*SEQ + sp] =
              (_Float16)(acc[mt][nt][r] + bn);
        }
      } else {                            // QK: RoPE then write [B,H,S,64]
        const int f = n & (DIM - 1);
        const int h = f >> 6, d = f & 63;
        const float invf = __powf(10000.0f, -(float)(d & 31) * (1.0f/32.0f));
        _Float16* dst = (n < DIM) ? qb : kb;
        #pragma unroll
        for (int r = 0; r < 8; ++r) {
          const int m = m0 + wv*32 + mt*16 + half*8 + r;
          const int bi = m >> 11, sp = m & (SEQ - 1);
          const float val = acc[mt][nt][r] + bn;
          // partner feature n^1 lives in lane^1 (same row)
          const float par = __shfl_xor(val, 1, 32);
          const float rot = (d & 1) ? par : -par;   // rotate_every_two
          const float ang = (float)sp * invf;
          const float o = val * __cosf(ang) + rot * __sinf(ang);
          dst[(((size_t)bi*HEADS + h)*SEQ + sp)*HDIM + d] = (_Float16)o;
        }
      }
    }
  }
}

// ---------------------------------------------------------------------------
// Kernel 2: flash attention.  4 waves/block handle 4 consecutive 16-query
// tiles of the SAME (b,h), so K/V tiles are staged once per block into LDS
// (async global->LDS on gfx1250).  S^T = K*Q^T (keys on M so softmax reduces
// in-lane), online softmax, then O^T += V^T * P^T.  8 WMMAs per 32-key step.
// ---------------------------------------------------------------------------
__global__ __launch_bounds__(128) void attn_kernel(
    const _Float16* __restrict__ qb, const _Float16* __restrict__ kb,
    const _Float16* __restrict__ vT, float* __restrict__ attnb)
{
  __shared__ _Float16 Ks[32][72];       // K tile [key][d], padded (144B rows)
  __shared__ _Float16 Vs[64][40];       // V^T tile [d][key], padded (80B rows)
  __shared__ _Float16 pT[4][16][40];    // per-wave P^T staging [q][key]

  const int tid  = threadIdx.x;
  const int wv   = tid >> 5;
  const int lane = tid & 31;
  const int half = lane >> 4;
  const int l16  = lane & 15;
  const int tile = blockIdx.x * 4 + wv;     // 0 .. B*H*(S/16)-1
  const int qt = tile & 127;                // query tile within (b,h)
  const int bh = blockIdx.x >> 5;           // 32 blocks per (b,h)

  // Q^T B-operand fragments: lane col q=l16, d = {half*8+0..7, half*8+16..23} (held all loop)
  const _Float16* qrow = qb + ((size_t)bh*SEQ + qt*16 + l16) * HDIM;
  const v16h b_q0 = hcat(*(const v8h*)(qrow + half*8),
                         *(const v8h*)(qrow + half*8 + 16));
  const v16h b_q1 = hcat(*(const v8h*)(qrow + 32 + half*8),
                         *(const v8h*)(qrow + 32 + half*8 + 16));

  const _Float16* kbase = kb + (size_t)bh * SEQ * HDIM;
  const _Float16* vbase = vT + (size_t)bh * HDIM * SEQ;

  v8f c_o[4] = {};
  float mrun = -1e30f, lrun = 0.0f;

  for (int kb0 = 0; kb0 < SEQ; kb0 += 32) {
    // ---- cooperative staging: K 32x64 (4KB) + V^T 64x32 (4KB), 16B/lane ----
    #pragma unroll
    for (int u = 0; u < 2; ++u) {
      const int c = tid * 2 + u;            // 0..255
      const int krow = c >> 3, kseg = c & 7;
      async_copy16(kbase + (size_t)(kb0 + krow)*HDIM + kseg*8,
                   &Ks[krow][kseg*8]);
      const int vrow = c >> 2, vseg = c & 3;
      async_copy16(vbase + (size_t)vrow*SEQ + kb0 + vseg*8,
                   &Vs[vrow][vseg*8]);
    }
    wait_async();
    __syncthreads();

    // ---- S^T tiles: 16 keys x 16 queries, K-dim = head dim (2 x 32) ----
    v8f c_s[2];
    #pragma unroll
    for (int j = 0; j < 2; ++j) {
      v16h ak0 = hcat(*(const v8h*)&Ks[j*16 + l16][half*8],
                      *(const v8h*)&Ks[j*16 + l16][half*8 + 16]);
      v16h ak1 = hcat(*(const v8h*)&Ks[j*16 + l16][32 + half*8],
                      *(const v8h*)&Ks[j*16 + l16][32 + half*8 + 16]);
      v8f z = {};
      c_s[j] = wmma16(ak0, b_q0, z);
      c_s[j] = wmma16(ak1, b_q1, c_s[j]);
    }
    // ---- online softmax (per query = per lane column) ----
    float mloc = -1e30f;
    #pragma unroll
    for (int j = 0; j < 2; ++j)
      #pragma unroll
      for (int r = 0; r < 8; ++r) {
        const float sv = c_s[j][r] * 0.125f;   // 1/sqrt(64)
        c_s[j][r] = sv;
        mloc = fmaxf(mloc, sv);
      }
    mloc = fmaxf(mloc, __shfl_xor(mloc, 16, 32));  // combine half-waves
    const float mnew  = fmaxf(mrun, mloc);
    const float alpha = __expf(mrun - mnew);
    float psum = 0.0f;
    #pragma unroll
    for (int j = 0; j < 2; ++j)
      #pragma unroll
      for (int r = 0; r < 8; r += 2) {
        const float p0 = __expf(c_s[j][r]     - mnew);
        const float p1 = __expf(c_s[j][r + 1] - mnew);
        psum += p0 + p1;
        v2h pp; pp.x = (_Float16)p0; pp.y = (_Float16)p1;
        *(v2h*)&pT[wv][l16][j*16 + half*8 + r] = pp;  // key = j*16+half*8+r
      }
    psum += __shfl_xor(psum, 16, 32);
    lrun = lrun * alpha + psum;
    mrun = mnew;

    // same-wave LDS RAW: DS ops are in-order, wait for visibility anyway
    asm volatile("s_wait_dscnt 0" ::: "memory");

    // P^T B-operand: lane col q=l16, keys {half*8+0..7, half*8+16..23}
    const v16h b_p = hcat(*(const v8h*)&pT[wv][l16][half*8],
                          *(const v8h*)&pT[wv][l16][half*8 + 16]);
    // ---- O^T += V^T * P^T ----
    #pragma unroll
    for (int dt = 0; dt < 4; ++dt) {
      v16h av = hcat(*(const v8h*)&Vs[dt*16 + l16][half*8],
                     *(const v8h*)&Vs[dt*16 + l16][half*8 + 16]);
      #pragma unroll
      for (int r = 0; r < 8; ++r) c_o[dt][r] *= alpha;
      c_o[dt] = wmma16(av, b_p, c_o[dt]);
    }
    __syncthreads();   // all waves done with Ks/Vs before restaging
  }

  // ---- finalize: divide by l, write fp32 [B,S,D] ----
  const float inv = 1.0f / lrun;
  const int token = qt*16 + l16;
  const int bi = bh >> 4, h = bh & (HEADS - 1);
  float* orow = attnb + ((size_t)bi*SEQ + token) * DIM + h * HDIM;
  #pragma unroll
  for (int dt = 0; dt < 4; ++dt) {
    float4 lo, hi;
    lo.x = c_o[dt][0]*inv; lo.y = c_o[dt][1]*inv;
    lo.z = c_o[dt][2]*inv; lo.w = c_o[dt][3]*inv;
    hi.x = c_o[dt][4]*inv; hi.y = c_o[dt][5]*inv;
    hi.z = c_o[dt][6]*inv; hi.w = c_o[dt][7]*inv;
    *(float4*)(orow + dt*16 + half*8)     = lo;
    *(float4*)(orow + dt*16 + half*8 + 4) = hi;
  }
}

// ---------------------------------------------------------------------------
// Kernel 3: LayerNorm over D=1024 (population variance, eps=1e-5)
// ---------------------------------------------------------------------------
__global__ __launch_bounds__(256) void ln_kernel(
    const float* __restrict__ attnb, const float* __restrict__ gamma,
    const float* __restrict__ beta, float* __restrict__ out)
{
  const int row = blockIdx.x;
  const int tid = threadIdx.x;
  const float* x = attnb + (size_t)row * DIM;
  float v[4], s = 0.0f, s2 = 0.0f;
  #pragma unroll
  for (int i = 0; i < 4; ++i) {
    v[i] = x[tid + i*256];
    s += v[i]; s2 += v[i]*v[i];
  }
  #pragma unroll
  for (int off = 16; off > 0; off >>= 1) {
    s  += __shfl_xor(s,  off, 32);
    s2 += __shfl_xor(s2, off, 32);
  }
  __shared__ float red[2][8];
  if ((tid & 31) == 0) { red[0][tid >> 5] = s; red[1][tid >> 5] = s2; }
  __syncthreads();
  float ts = 0.0f, ts2 = 0.0f;
  #pragma unroll
  for (int w = 0; w < 8; ++w) { ts += red[0][w]; ts2 += red[1][w]; }
  const float mu   = ts  * (1.0f / DIM);
  const float var  = ts2 * (1.0f / DIM) - mu * mu;
  const float rstd = rsqrtf(var + 1e-5f);
  float* o = out + (size_t)row * DIM;
  #pragma unroll
  for (int i = 0; i < 4; ++i) {
    const int c = tid + i*256;
    o[c] = (v[i] - mu) * rstd * gamma[c] + beta[c];
  }
}

// ---------------------------------------------------------------------------
extern "C" void kernel_launch(void* const* d_in, const int* in_sizes, int n_in,
                              void* d_out, int out_size, void* d_ws, size_t ws_size,
                              hipStream_t stream) {
  (void)in_sizes; (void)n_in; (void)out_size; (void)ws_size;
  const float* x_qk  = (const float*)d_in[0];
  const float* x_v   = (const float*)d_in[1];
  const float* W_qk  = (const float*)d_in[2];
  const float* b_qk  = (const float*)d_in[3];
  const float* W_v   = (const float*)d_in[4];
  const float* b_v   = (const float*)d_in[5];
  const float* gamma = (const float*)d_in[6];
  const float* beta  = (const float*)d_in[7];

  // workspace: q,k [B,H,S,64] f16 + vT [B,H,64,S] f16 + attn [B,S,D] f32 = ~42MB
  const size_t qkv = (size_t)BATCH * HEADS * SEQ * HDIM;   // 4,194,304
  _Float16* qbuf = (_Float16*)d_ws;
  _Float16* kbuf = qbuf + qkv;
  _Float16* vTb  = kbuf + qkv;
  float*    attn = (float*)(vTb + qkv);

  // QK projection + RoPE (N=2048), V projection (N=1024, transposed store)
  proj_rope_kernel<<<dim3(2*DIM/64, BATCH*SEQ/128), 128, 0, stream>>>(
      x_qk, W_qk, b_qk, qbuf, kbuf, vTb, 2*DIM, 0);
  proj_rope_kernel<<<dim3(DIM/64, BATCH*SEQ/128), 128, 0, stream>>>(
      x_v, W_v, b_v, qbuf, kbuf, vTb, DIM, 1);
  // flash attention: B*H*(S/16)=4096 wave-tiles, 4 waves/block
  attn_kernel<<<BATCH*HEADS*(SEQ/16)/4, 128, 0, stream>>>(qbuf, kbuf, vTb, attn);
  // layernorm: one block per token row
  ln_kernel<<<BATCH*SEQ, 256, 0, stream>>>(attn, gamma, beta, (float*)d_out);
}
